// clusteringEMA_v2_torch_73237782331476
// MI455X (gfx1250) — matrix operand order repaired and checked by reading, compiled, not run on hardware
//
#include <hip/hip_runtime.h>
#include <hip/hip_bf16.h>

#define GAMMA_ 0.99f
#define EPS_   1e-5f
#define Bsz 65536
#define Dsz 512
#define Ksz 1024
#define Csz 100

typedef __bf16 bf16;
typedef __attribute__((ext_vector_type(16))) __bf16 v16bf;
typedef __attribute__((ext_vector_type(8)))  __bf16 v8bf;
typedef __attribute__((ext_vector_type(8)))  float  v8f;

__device__ __forceinline__ void bf_split(float x, bf16& h, bf16& l) {
  h = (bf16)x;                 // RNE
  l = (bf16)(x - (float)h);    // residual
}

// A-fragment (16x32 bf16, ISA 7.12.2): lane<16: j0..7->K0..7, j8..15->K16..23
//                                      lane>=16: j0..7->K8..15, j8..15->K24..31
// caller passes p already offset by (lane<16 ? 0 : 8); halves at +0 and +16.
__device__ __forceinline__ v16bf load_frag_a(const bf16* p) {
  v8bf a = *(const v8bf*)(p);
  v8bf b = *(const v8bf*)(p + 16);
  v16bf r;
#pragma unroll
  for (int j = 0; j < 8; ++j) { r[j] = a[j]; r[j + 8] = b[j]; }
  return r;
}

// B-fragment (32x16 bf16): lane holds col n = lane%16; lanes 0-15 K=0..7,16..23,
// lanes 16-31 K=8..15,24..31 -> halves at +0 and +16 (same as A).
__device__ __forceinline__ v16bf load_frag_b(const bf16* p) {
  v8bf a = *(const v8bf*)(p);
  v8bf b = *(const v8bf*)(p + 16);
  v16bf r;
#pragma unroll
  for (int j = 0; j < 8; ++j) { r[j] = a[j]; r[j + 8] = b[j]; }
  return r;
}

// ---------------- prep: W -> bf16 hi/lo, column-major; and w_sq -------------
__global__ void k_prep_w(const float* __restrict__ W, bf16* __restrict__ Wh,
                         bf16* __restrict__ Wl) {
  int i = blockIdx.x * blockDim.x + threadIdx.x;  // over D*K, coalesced read
  int d = i >> 10;
  int k = i & (Ksz - 1);
  float v = W[i];
  bf16 h, l; bf_split(v, h, l);
  Wh[(size_t)k * Dsz + d] = h;
  Wl[(size_t)k * Dsz + d] = l;
}

__global__ void k_wsq(const float* __restrict__ W, float* __restrict__ wsq) {
  int k = blockIdx.x * blockDim.x + threadIdx.x;
  float s = 0.f;
  for (int d = 0; d < Dsz; ++d) { float v = W[(size_t)d * Ksz + k]; s = fmaf(v, v, s); }
  wsq[k] = s;
}

// ---------------- fused distance GEMM + argmin ------------------------------
#define MTILE 64
#define MSUB  4
#define APAD  8
#define AROW  (Dsz + APAD)   // 520 bf16 per row (bank-conflict pad)

// al_off: runtime value == MTILE*AROW (elements). Passed as an argument so the
// compiler keeps a DISTINCT base VGPR for the lo-tile and all ds_load offsets
// stay within the 16-bit immediate field (no per-iteration v_add + WAR nops).
__global__ __launch_bounds__(256) void k_dist_argmin(
    const float* __restrict__ X, const bf16* __restrict__ Wh,
    const bf16* __restrict__ Wl, const float* __restrict__ wsq,
    int* __restrict__ argmin_ws, float* __restrict__ argmin_out, int al_off) {
  extern __shared__ __align__(16) char smem[];
  bf16* Ah = (bf16*)smem;
  bf16* Al = Ah + MTILE * AROW;
  unsigned long long* rowmin = (unsigned long long*)(Al + MTILE * AROW);

  const int tid   = threadIdx.x;
  const int wave  = tid >> 5;
  const int lane  = tid & 31;
  const int row16 = lane & 15;
  const int aoff  = (lane < 16) ? 0 : 8;    // A fragment K-offset pattern
  const int boff  = (lane < 16) ? 0 : 8;    // B fragment K-offset pattern
  const int m0    = blockIdx.x * MTILE;

  // Stage A tile as bf16 hi/lo into LDS (float4 global reads, coalesced)
  for (int i = tid; i < MTILE * (Dsz / 4); i += 256) {
    int r = i >> 7;              // Dsz/4 == 128
    int c = (i & 127) << 2;
    const float4 v = *(const float4*)(X + (size_t)(m0 + r) * Dsz + c);
    bf16 h, l;
    bf_split(v.x, h, l); Ah[r * AROW + c + 0] = h; Al[r * AROW + c + 0] = l;
    bf_split(v.y, h, l); Ah[r * AROW + c + 1] = h; Al[r * AROW + c + 1] = l;
    bf_split(v.z, h, l); Ah[r * AROW + c + 2] = h; Al[r * AROW + c + 2] = l;
    bf_split(v.w, h, l); Ah[r * AROW + c + 3] = h; Al[r * AROW + c + 3] = l;
  }
  if (tid < MTILE) rowmin[tid] = ~0ULL;
  __syncthreads();

  // two independent per-lane bases (al_off is opaque -> not CSE'd together)
  const bf16* ahp = Ah + row16 * AROW + aoff;
  const bf16* alp = Ah + row16 * AROW + aoff + al_off;

  unsigned long long mk[MSUB][8];
#pragma unroll
  for (int s = 0; s < MSUB; ++s)
#pragma unroll
    for (int r = 0; r < 8; ++r) mk[s][r] = ~0ULL;

  const v8f vzero = {0.f, 0.f, 0.f, 0.f, 0.f, 0.f, 0.f, 0.f};

  for (int n0 = 0; n0 < Ksz; n0 += 128) {
    const int n = n0 + wave * 16 + row16;
    const bf16* whc = Wh + (size_t)n * Dsz;
    const bf16* wlc = Wl + (size_t)n * Dsz;

    // warm L2/WGP$ for next chunk's columns (global_prefetch_b8)
    if (n0 + 128 < Ksz) {
      __builtin_prefetch(whc + (size_t)128 * Dsz, 0, 1);
      __builtin_prefetch(whc + (size_t)128 * Dsz + 256, 0, 1);
      __builtin_prefetch(wlc + (size_t)128 * Dsz, 0, 1);
      __builtin_prefetch(wlc + (size_t)128 * Dsz + 256, 0, 1);
    }

    v8f acc[MSUB];
#pragma unroll
    for (int s = 0; s < MSUB; ++s) acc[s] = vzero;

#pragma unroll 4
    for (int dk = 0; dk < Dsz; dk += 32) {
      v16bf bh = load_frag_b(whc + dk + boff);
      v16bf bl = load_frag_b(wlc + dk + boff);
      v16bf af[MSUB];   // hi A-fragments
      v16bf ag[MSUB];   // lo A-fragments (separate regs: no WAR on in-flight WMMA)
#pragma unroll
      for (int s = 0; s < MSUB; ++s) af[s] = load_frag_a(ahp + s * 16 * AROW + dk);
#pragma unroll
      for (int s = 0; s < MSUB; ++s) ag[s] = load_frag_a(alp + s * 16 * AROW + dk);
      // three passes, s-interleaved: dependence distance 4 per accumulator
#pragma unroll
      for (int s = 0; s < MSUB; ++s)
        acc[s] = __builtin_amdgcn_wmma_f32_16x16x32_bf16(false, af[s], false, bh,
                                                         (short)0, acc[s], false, false);
#pragma unroll
      for (int s = 0; s < MSUB; ++s)
        acc[s] = __builtin_amdgcn_wmma_f32_16x16x32_bf16(false, af[s], false, bl,
                                                         (short)0, acc[s], false, false);
#pragma unroll
      for (int s = 0; s < MSUB; ++s)
        acc[s] = __builtin_amdgcn_wmma_f32_16x16x32_bf16(false, ag[s], false, bh,
                                                         (short)0, acc[s], false, false);
    }

    // d2 (minus the row-constant ||x||^2, irrelevant for argmin): wsq[n] - 2*dot
    const float wq = wsq[n];
#pragma unroll
    for (int s = 0; s < MSUB; ++s)
#pragma unroll
      for (int r = 0; r < 8; ++r) {
        float f = fmaf(-2.0f, acc[s][r], wq);
        unsigned u = __float_as_uint(f);
        u = (u & 0x80000000u) ? ~u : (u | 0x80000000u);  // monotone float->uint
        unsigned long long key = ((unsigned long long)u << 32) | (unsigned)n;
        if (key < mk[s][r]) mk[s][r] = key;
      }
  }

  // cross-lane / cross-wave reduction via LDS u64 min atomics
#pragma unroll
  for (int s = 0; s < MSUB; ++s)
#pragma unroll
    for (int r = 0; r < 8; ++r) {
      int row = s * 16 + r + ((lane >= 16) ? 8 : 0);  // C/D layout (ISA 7.12.2)
      atomicMin(&rowmin[row], mk[s][r]);
    }
  __syncthreads();
  if (tid < MTILE) {
    unsigned idx = (unsigned)(rowmin[tid] & 0xFFFFFFFFu);
    argmin_ws[m0 + tid]  = (int)idx;
    argmin_out[m0 + tid] = (float)idx;
  }
}

// ---------------- scatter phase ---------------------------------------------
__global__ void k_count_hist(const int* __restrict__ am, const int* __restrict__ keys,
                             float* __restrict__ counts, float* __restrict__ bh) {
  int b = blockIdx.x * blockDim.x + threadIdx.x;
  int k = am[b];
  atomicAdd(&counts[k], 1.0f);
  atomicAdd(&bh[(size_t)k * Csz + keys[b]], 1.0f);
}

__global__ void k_scatter_embed(const float* __restrict__ X, const int* __restrict__ am,
                                float* __restrict__ esum) {
  int i = blockIdx.x * blockDim.x + threadIdx.x;  // over B * D/4
  int b = i >> 7;
  int j = (i & 127) << 2;
  int k = am[b];
  const float4 v = *(const float4*)(X + (size_t)b * Dsz + j);
  atomicAdd(&esum[(size_t)(j + 0) * Ksz + k], v.x);
  atomicAdd(&esum[(size_t)(j + 1) * Ksz + k], v.y);
  atomicAdd(&esum[(size_t)(j + 2) * Ksz + k], v.z);
  atomicAdd(&esum[(size_t)(j + 3) * Ksz + k], v.w);
}

// ---------------- EMA finalize ----------------------------------------------
__global__ void k_finalize_cs(const float* __restrict__ counts,
                              const float* __restrict__ cs_in,
                              float* __restrict__ out_cs, float* __restrict__ nsum) {
  __shared__ float red[Ksz];
  int k = threadIdx.x;
  float c = counts[k];
  float nic = (c == 0.f) ? 1.f : c;
  float ncs = cs_in[k] * GAMMA_ + (1.f - GAMMA_) * nic;
  out_cs[k] = ncs;
  red[k] = ncs;
  __syncthreads();
  for (int s = Ksz / 2; s > 0; s >>= 1) {
    if (k < s) red[k] += red[k + s];
    __syncthreads();
  }
  if (k == 0) *nsum = red[0];
}

__global__ void k_finalize_w(const float* __restrict__ esum, const float* __restrict__ ea,
                             const float* __restrict__ ncs, const float* __restrict__ nsum,
                             float* __restrict__ out_ea, float* __restrict__ out_w) {
  int i = blockIdx.x * blockDim.x + threadIdx.x;  // over D*K, k fastest
  int k = i & (Ksz - 1);
  float e = ea[i] * GAMMA_ + (1.f - GAMMA_) * esum[i];
  out_ea[i] = e;
  float n = *nsum;
  float cn = (ncs[k] + EPS_) / (n + (float)Ksz * EPS_) * n;
  out_w[i] = e / cn;
}

__global__ void k_finalize_hist(const float* __restrict__ hist, const float* __restrict__ bh,
                                float* __restrict__ out_h) {
  int i = blockIdx.x * blockDim.x + threadIdx.x;  // over K*C
  out_h[i] = hist[i] * GAMMA_ + (1.f - GAMMA_) * bh[i];
}

// ---------------- launch ----------------------------------------------------
extern "C" void kernel_launch(void* const* d_in, const int* in_sizes, int n_in,
                              void* d_out, int out_size, void* d_ws, size_t ws_size,
                              hipStream_t stream) {
  (void)in_sizes; (void)n_in; (void)out_size; (void)ws_size;
  const float* X    = (const float*)d_in[0];
  const int*   keys = (const int*)d_in[1];
  const float* W    = (const float*)d_in[2];
  const float* cs   = (const float*)d_in[3];
  const float* ea   = (const float*)d_in[4];
  const float* hist = (const float*)d_in[5];

  float* out    = (float*)d_out;
  float* out_w  = out;                               // D*K
  float* out_cs = out_w + (size_t)Dsz * Ksz;         // K
  float* out_ea = out_cs + Ksz;                      // D*K
  float* out_h  = out_ea + (size_t)Dsz * Ksz;        // K*C
  float* out_am = out_h + (size_t)Ksz * Csz;         // B

  char* ws = (char*)d_ws;
  bf16*  Wh     = (bf16*)ws;                                        // 1 MB
  bf16*  Wl     = (bf16*)(ws + (size_t)Ksz * Dsz * 2);              // 1 MB
  float* wsq    = (float*)(ws + (size_t)Ksz * Dsz * 4);             // 4 KB
  int*   am     = (int*)((char*)wsq + Ksz * 4);                     // 256 KB
  float* counts = (float*)((char*)am + (size_t)Bsz * 4);            // 4 KB
  float* esum   = (float*)((char*)counts + Ksz * 4);                // 2 MB
  float* bhist  = (float*)((char*)esum + (size_t)Dsz * Ksz * 4);    // 400 KB
  float* nsum   = (float*)((char*)bhist + (size_t)Ksz * Csz * 4);   // 4 B

  hipMemsetAsync(counts, 0, Ksz * 4, stream);
  hipMemsetAsync(esum, 0, (size_t)Dsz * Ksz * 4, stream);
  hipMemsetAsync(bhist, 0, (size_t)Ksz * Csz * 4, stream);

  k_prep_w<<<(Dsz * Ksz) / 256, 256, 0, stream>>>(W, Wh, Wl);
  k_wsq<<<Ksz / 256, 256, 0, stream>>>(W, wsq);

  size_t shmem = (size_t)MTILE * AROW * 2 * 2 + MTILE * 8;  // Ah + Al + rowmin
  k_dist_argmin<<<Bsz / MTILE, 256, shmem, stream>>>(X, Wh, Wl, wsq, am, out_am,
                                                     MTILE * AROW);

  k_count_hist<<<Bsz / 256, 256, 0, stream>>>(am, keys, counts, bhist);
  k_scatter_embed<<<(size_t)Bsz * (Dsz / 4) / 256, 256, 0, stream>>>(X, am, esum);

  k_finalize_cs<<<1, Ksz, 0, stream>>>(counts, cs, out_cs, nsum);
  k_finalize_w<<<(Dsz * Ksz) / 256, 256, 0, stream>>>(esum, ea, out_cs, nsum, out_ea, out_w);
  k_finalize_hist<<<(Ksz * Csz) / 256, 256, 0, stream>>>(hist, bhist, out_h);
}
